// SynthesisBlock_59476707115523
// MI455X (gfx1250) — compile-verified, matrix operand order checked by reading, not run on hardware
//
#include <hip/hip_runtime.h>
#include <hip/hip_bf16.h>
#include <math.h>

typedef _Float16 v16h __attribute__((ext_vector_type(16)));
typedef _Float16 v8h  __attribute__((ext_vector_type(8)));
typedef _Float16 v4h  __attribute__((ext_vector_type(4)));
typedef float    v8f  __attribute__((ext_vector_type(8)));

#define B_    8
#define CIN_  512
#define COUT_ 512
#define RES_  64
#define LSTRIDE 40                  // f16 per (row,col) entry: 32 cin + 8 pad
#define BUFELEMS (6 * 66 * LSTRIDE) // f16 per LDS buffer (15840)

// ---------------------------------------------------------------------------
// 1) styles[b,c] = m1*m2 + m3 from the style FC (weight scaled 1/sqrt(512))
// ---------------------------------------------------------------------------
__global__ void styles_kernel(const float* __restrict__ w,
                              const float* __restrict__ aW,
                              const float* __restrict__ ab,
                              float* __restrict__ styles) {
  int i = blockIdx.x * 256 + threadIdx.x;          // 0..4095
  int b = i >> 9, c = i & 511;
  const float* wb = w + b * 512;
  const float* r1 = aW + (size_t)c * 512;
  const float* r2 = aW + (size_t)(c + 512) * 512;
  const float* r3 = aW + (size_t)(c + 1024) * 512;
  float a1 = 0.f, a2 = 0.f, a3 = 0.f;
  for (int k = 0; k < 512; ++k) {
    float wv = wb[k];
    a1 += wv * r1[k]; a2 += wv * r2[k]; a3 += wv * r3[k];
  }
  const float inv = 0.044194173824159216f;         // 1/sqrt(512)
  float m1 = a1 * inv + ab[c];
  float m2 = a2 * inv + ab[c + 512];
  float m3 = a3 * inv + ab[c + 1024];
  styles[i] = m1 * m2 + m3;
}

// ---------------------------------------------------------------------------
// 2) GroupNorm (fp32) -> xn_f32 NCHW (residual path) + xn_f16 NHWC (conv path)
// ---------------------------------------------------------------------------
__global__ void gn_kernel(const float* __restrict__ x,
                          const float* __restrict__ gnw,
                          const float* __restrict__ gnb,
                          float* __restrict__ xnf,
                          _Float16* __restrict__ xnh) {
  const int g = blockIdx.x, b = blockIdx.y, tid = threadIdx.x;
  const float* xb = x + ((size_t)(b * CIN_ + g * 16) << 12);
  float s = 0.f, ss = 0.f;
  for (int e = tid; e < 16 * 4096; e += 256) {
    float v = xb[e]; s += v; ss += v * v;
  }
  __shared__ float rs[256], rss[256];
  rs[tid] = s; rss[tid] = ss; __syncthreads();
  for (int o = 128; o > 0; o >>= 1) {
    if (tid < o) { rs[tid] += rs[tid + o]; rss[tid] += rss[tid + o]; }
    __syncthreads();
  }
  const float mean = rs[0] * (1.f / 65536.f);
  const float var  = rss[0] * (1.f / 65536.f) - mean * mean;
  const float invs = rsqrtf(var + 1e-5f);
  for (int e = tid; e < 16 * 4096; e += 256) {
    int c = g * 16 + (e >> 12);
    int p = e & 4095;
    float v = (xb[e] - mean) * invs * gnw[c] + gnb[c];
    xnf[(((size_t)(b * CIN_ + c)) << 12) + p] = v;                 // NCHW f32
    xnh[((((size_t)b << 12) + p) << 9) + c] = (_Float16)v;         // NHWC f16
  }
}

// ---------------------------------------------------------------------------
// 3) demod[b,cout] = rsqrt( sum_cin styles^2 * sum_k weight^2 + 1e-8 )
// ---------------------------------------------------------------------------
__global__ void demod_kernel(const float* __restrict__ weight,
                             const float* __restrict__ styles,
                             float* __restrict__ demod) {
  const int co = blockIdx.x, b = blockIdx.y, tid = threadIdx.x;
  float s = 0.f;
  for (int cin = tid; cin < CIN_; cin += 256) {
    const float* wp = weight + (((size_t)co << 9) + cin) * 9;
    float w9 = 0.f;
#pragma unroll
    for (int k = 0; k < 9; ++k) w9 += wp[k] * wp[k];
    float st = styles[(b << 9) + cin];
    s += w9 * st * st;
  }
  __shared__ float red[256];
  red[tid] = s; __syncthreads();
  for (int o = 128; o > 0; o >>= 1) {
    if (tid < o) red[tid] += red[tid + o];
    __syncthreads();
  }
  if (tid == 0) demod[(b << 9) + co] = rsqrtf(red[0] + 1e-8f);
}

// ---------------------------------------------------------------------------
// 4) wmod[b][tap][cout][cin] (f16) = weight * style * demod  (GEMM-A layout)
// ---------------------------------------------------------------------------
__global__ void wmod_kernel(const float* __restrict__ weight,
                            const float* __restrict__ styles,
                            const float* __restrict__ demod,
                            _Float16* __restrict__ wmod) {
  size_t i = (size_t)blockIdx.x * 256 + threadIdx.x;   // < 8*9*512*512
  int cin  = (int)(i & 511);
  int cout = (int)((i >> 9) & 511);
  int rest = (int)(i >> 18);                           // b*9 + t
  int t = rest % 9;
  int b = rest / 9;
  float v = weight[((((size_t)cout << 9) + cin) * 9) + t]
          * styles[(b << 9) + cin] * demod[(b << 9) + cout];
  wmod[i] = (_Float16)v;
}

// ---------------------------------------------------------------------------
// 5) Implicit-GEMM 3x3 conv: V_WMMA_F32_16X16X32_F16 with double-buffered
//    GLOBAL_LOAD_ASYNC_TO_LDS staging (ASYNCcnt software pipeline),
//    bank-conflict-free padded LDS, fully unrolled taps, fused epilogue.
//    Block: (b, 128 couts, 4 output rows). 8 waves = 2 cout halves x 4 rows.
// ---------------------------------------------------------------------------
__global__ __launch_bounds__(256, 1) void conv_wmma_kernel(
    const _Float16* __restrict__ wmod,   // [B][9][COUT][CIN]
    const _Float16* __restrict__ xnh,    // [B][RES][RES][CIN] f16
    const float* __restrict__ xnf,       // [B][CIN][RES*RES] f32 (residual)
    const float* __restrict__ noise,     // [B][RES*RES]
    const float* __restrict__ bias,      // [COUT]
    const float* __restrict__ gamma,     // [COUT]
    const float* __restrict__ nstr,      // scalar
    float* __restrict__ out)             // [B][COUT][RES*RES]
{
  // 2 buffers x rows(6) x cols+halo(66) x LSTRIDE(40) f16 = 63360 B
  __shared__ __align__(16) _Float16 lds[2 * BUFELEMS];
  const int tid  = threadIdx.x;
  const int lane = tid & 31;
  const int wave = tid >> 5;
  const int wm   = wave & 1;             // cout half of the 128-block
  const int wn   = wave >> 1;            // output row within the 4-row block
  const int rowblk  = blockIdx.x;        // 0..15  (4 rows each)
  const int coutblk = blockIdx.y;        // 0..3   (128 couts each)
  const int b       = blockIdx.z;        // 0..7
  const int row      = rowblk * 4 + wn;
  const int coutbase = coutblk * 128 + wm * 64;
  const int half = lane >> 4;            // K-interleave half per ISA layout
  const int l15  = lane & 15;            // M (A) / N (B,C,D) index

  // Zero both buffers once: halo/OOB entries are chunk-invariant and never
  // overwritten (masked issues are redirected to the pad dump slot).
  {
    v4h z4 = {};
    for (int e = tid; e < 2 * BUFELEMS / 4; e += 256)
      ((v4h*)lds)[e] = z4;
  }
  __syncthreads();

  // Per-thread chunk-invariant staging descriptors: ciq fixed, rest steps 32.
  const int ciq   = tid & 7;             // which 4-f16 quad of the 32-cin line
  const int rest0 = tid >> 3;            // 0..31

  // Stage one 32-cin chunk into buffer `sel`. Exactly 13 async issues per
  // thread (uniform per wave) so s_wait_asynccnt immediates are exact;
  // invalid entries go to a never-read dump slot with clamped addresses.
  auto stage = [&](int cin0, int sel) {
    int col = rest0, ry = 0;             // rest = rest0 + 32*it, col=rest%66, ry=rest/66
#pragma unroll
    for (int it = 0; it < 13; ++it) {
      const int y = rowblk * 4 - 1 + ry;
      const int x = col - 1;
      const bool tail  = (it == 12) && (rest0 >= 12);      // e >= 3168
      const bool valid = !tail && ((unsigned)y < 64u) && ((unsigned)x < 64u);
      unsigned lofs = valid
          ? (unsigned)((sel * BUFELEMS + (ry * 66 + col) * LSTRIDE + (ciq << 2)) * 2)
          : (unsigned)((sel * BUFELEMS + 32) * 2);         // pad dump slot
      const int yc = y < 0 ? 0 : (y > 63 ? 63 : y);
      const int xc = x < 0 ? 0 : (x > 63 ? 63 : x);
      unsigned long long gaddr = (unsigned long long)(xnh +
          ((((size_t)(b * 64 + yc) * 64 + xc) << 9) + cin0 + (ciq << 2)));
      asm volatile("global_load_async_to_lds_b64 %0, %1, off"
                   :: "v"(lofs), "v"(gaddr) : "memory");
      col += 32; if (col >= 66) { col -= 66; ry += 1; }
    }
  };

  v8f zero = {};
  v8f acc[4][4];
#pragma unroll
  for (int i = 0; i < 4; ++i)
#pragma unroll
    for (int j = 0; j < 4; ++j) acc[i][j] = zero;

  stage(0, 0);                           // prologue: batch 0 -> buffer 0

  for (int c = 0; c < 16; ++c) {
    const int cin0 = c << 5;
    if (c < 15) {
      stage(cin0 + 32, (c + 1) & 1);     // overlap: batch c+1 -> other buffer
      // Async loads complete in order: allowing the 13 just-issued ops to
      // remain outstanding guarantees batch c has fully landed in LDS.
      asm volatile("s_wait_asynccnt 0xd" ::: "memory");
    } else {
      asm volatile("s_wait_asynccnt 0x0" ::: "memory");
    }
    __syncthreads();                     // batch c visible to all waves

    const _Float16* buf = lds + (c & 1) * BUFELEMS;
#pragma unroll
    for (int t = 0; t < 9; ++t) {
      const int ky = t / 3, kx = t - 3 * ky;
      // A fragments: 16x32 f16, lane half selects K-interleave (2 x b128).
      v16h A[4];
#pragma unroll
      for (int mt = 0; mt < 4; ++mt) {
        const int co = coutbase + mt * 16 + l15;
        const _Float16* ap = wmod +
            (((((size_t)(b * 9 + t)) << 9) + co) << 9) + cin0 + (half << 3);
        v8h lo = *(const v8h*)ap;
        v8h hi = *(const v8h*)(ap + 16);
#pragma unroll
        for (int i = 0; i < 8; ++i) { A[mt][i] = lo[i]; A[mt][8 + i] = hi[i]; }
      }
#pragma unroll
      for (int nt = 0; nt < 4; ++nt) {
        // B fragment: 32x16 f16 from LDS (2 x ds_load_b128), shared by mtiles.
        const int bc = nt * 16 + l15 + kx;           // halo col = x + kx
        const int br = wn + ky;                      // halo row
        const _Float16* bp = buf + (br * 66 + bc) * LSTRIDE + (half << 3);
        v8h lo = *(const v8h*)bp;
        v8h hi = *(const v8h*)(bp + 16);
        v16h Bf;
#pragma unroll
        for (int i = 0; i < 8; ++i) { Bf[i] = lo[i]; Bf[8 + i] = hi[i]; }
#pragma unroll
        for (int mt = 0; mt < 4; ++mt)
          acc[mt][nt] = __builtin_amdgcn_wmma_f32_16x16x32_f16(
              false, A[mt], false, Bf, (short)0, acc[mt][nt], false, false);
      }
    }
    __syncthreads();                     // done reading buf before next stage
  }

  // Fused epilogue: noise, bias + lrelu * 2*sqrt(2), layer-scale residual.
  const float ns   = nstr[0];
  const float gain = 2.8284271247461903f;  // sqrt(2) * 2
  const float s2   = 1.4142135623730951f;  // sqrt(2)
#pragma unroll
  for (int nt = 0; nt < 4; ++nt) {
    const int x = nt * 16 + l15;
    const float nz = noise[((size_t)b << 12) + row * 64 + x] * ns;
#pragma unroll
    for (int mt = 0; mt < 4; ++mt) {
      const int cbase = coutbase + mt * 16 + half * 8;   // D layout: M=r+8*half
#pragma unroll
      for (int r = 0; r < 8; ++r) {
        const int co = cbase + r;
        float v = acc[mt][nt][r] + nz;
        v += bias[co];
        v = (v > 0.f ? v : 0.2f * v) * gain;
        const size_t oi = (((size_t)(b * COUT_ + co)) << 12) + row * 64 + x;
        out[oi] = (gamma[co] * v + xnf[oi]) * s2;
      }
    }
  }
}

// ---------------------------------------------------------------------------
extern "C" void kernel_launch(void* const* d_in, const int* in_sizes, int n_in,
                              void* d_out, int out_size, void* d_ws, size_t ws_size,
                              hipStream_t stream) {
  (void)in_sizes; (void)n_in; (void)out_size; (void)ws_size;
  const float* x         = (const float*)d_in[0];
  const float* w         = (const float*)d_in[1];
  const float* noise     = (const float*)d_in[2];
  const float* weight    = (const float*)d_in[3];
  const float* bias      = (const float*)d_in[4];
  const float* affine_W  = (const float*)d_in[5];
  const float* affine_b  = (const float*)d_in[6];
  const float* gn_weight = (const float*)d_in[7];
  const float* gn_bias   = (const float*)d_in[8];
  const float* gamma     = (const float*)d_in[9];
  const float* nstr      = (const float*)d_in[10];
  float* out = (float*)d_out;

  char* ws = (char*)d_ws;
  float*    styles = (float*)(ws);                                   // 16 KB
  float*    demod  = (float*)(ws + (16 << 10));                      // 16 KB
  float*    xnf    = (float*)(ws + (32 << 10));                      // 64 MB
  _Float16* xnh    = (_Float16*)(ws + (32 << 10) + (size_t)67108864);        // 32 MB
  _Float16* wmod   = (_Float16*)(ws + (32 << 10) + (size_t)67108864 + 33554432); // 36 MB

  styles_kernel<<<16, 256, 0, stream>>>(w, affine_W, affine_b, styles);
  gn_kernel<<<dim3(32, 8), 256, 0, stream>>>(x, gn_weight, gn_bias, xnf, xnh);
  demod_kernel<<<dim3(512, 8), 256, 0, stream>>>(weight, styles, demod);
  wmod_kernel<<<73728, 256, 0, stream>>>(weight, styles, demod, wmod);
  conv_wmma_kernel<<<dim3(16, 4, 8), 256, 0, stream>>>(
      wmod, xnh, xnf, noise, bias, gamma, nstr, out);
}
